// RFFKernel_45019847197425
// MI455X (gfx1250) — matrix operand order repaired and checked by reading, not compiled, not run on hardware
//
#include <hip/hip_runtime.h>
#include <hip/hip_bf16.h>
#include <math.h>

typedef __attribute__((ext_vector_type(16))) _Float16 v16h;
typedef __attribute__((ext_vector_type(8)))  _Float16 v8h;
typedef __attribute__((ext_vector_type(8)))  float    v8f;
typedef __attribute__((ext_vector_type(4)))  float    v4f;

#define NFEAT 256
#define DIN   64
#define PROWS 8
#define RFF_SCALE 0.08838834764831845f   /* sqrt(2/256) */

#define SEQ 2048
#define BM 128
#define BN 128
#define BK 32
#define LDL 40                 /* padded LDS stride in halves */
#define BUFBYTES (BM * LDL * 2) /* one tile buffer: 10240 bytes */

// ---------------------------------------------------------------------------
// Phase 1: phi(x) = normalize( SCALE * cos(W x + b) ), stored as f16.
// ---------------------------------------------------------------------------
__global__ void __launch_bounds__(256)
phi_kernel(const float* __restrict__ x, const float* __restrict__ Wm,
           const float* __restrict__ bvec, _Float16* __restrict__ out)
{
  __shared__ float xs[PROWS][DIN];
  __shared__ float fs[PROWS][NFEAT];
  __shared__ float inv_s[PROWS];

  const int t = threadIdx.x;
  const long rowbase = (long)blockIdx.x * PROWS;

  for (int i = t; i < PROWS * DIN; i += 256)
    xs[i >> 6][i & 63] = __builtin_nontemporal_load(&x[rowbase * DIN + i]);
  __syncthreads();

  const int m = t;
  float acc[PROWS];
#pragma unroll
  for (int r = 0; r < PROWS; ++r) acc[r] = 0.f;

  const float* wrow = Wm + m * DIN;
#pragma unroll
  for (int d = 0; d < DIN; d += 4) {
    v4f w = *(const v4f*)(wrow + d);
#pragma unroll
    for (int r = 0; r < PROWS; ++r) {
      acc[r] = fmaf(xs[r][d + 0], w[0], acc[r]);
      acc[r] = fmaf(xs[r][d + 1], w[1], acc[r]);
      acc[r] = fmaf(xs[r][d + 2], w[2], acc[r]);
      acc[r] = fmaf(xs[r][d + 3], w[3], acc[r]);
    }
  }

  const float bb = bvec[m];
#pragma unroll
  for (int r = 0; r < PROWS; ++r)
    fs[r][m] = RFF_SCALE * cosf(acc[r] + bb);
  __syncthreads();

  const int wid = t >> 5, lane = t & 31;
  float s = 0.f;
#pragma unroll
  for (int i = 0; i < NFEAT / 32; ++i) {
    float f = fs[wid][lane + i * 32];
    s = fmaf(f, f, s);
  }
#pragma unroll
  for (int off = 16; off > 0; off >>= 1)
    s += __shfl_xor(s, off, 32);
  if (lane == 0) inv_s[wid] = 1.0f / (sqrtf(s) + 1e-8f);
  __syncthreads();

#pragma unroll
  for (int r = 0; r < PROWS; ++r)
    out[(rowbase + r) * NFEAT + m] = (_Float16)(fs[r][m] * inv_s[r]);
}

// ---------------------------------------------------------------------------
// CDNA5 async global->LDS copy (16B/lane), tracked by ASYNCcnt.
// Generic LDS pointer truncated to 32 bits == LDS byte offset (aperture rule).
// ---------------------------------------------------------------------------
__device__ __forceinline__ unsigned lds_off(const void* p) {
  return (unsigned)(unsigned long long)p;
}
__device__ __forceinline__ void async_cp16(unsigned lds, const void* gaddr) {
  asm volatile("global_load_async_to_lds_b128 %0, %1, off"
               :: "v"(lds), "v"(gaddr) : "memory");
}
__device__ __forceinline__ void wait_async_le4() {
  asm volatile("s_wait_asynccnt 0x4" ::: "memory");
}
__device__ __forceinline__ void wait_async_0() {
  asm volatile("s_wait_asynccnt 0x0" ::: "memory");
}

// ---------------------------------------------------------------------------
// Fragment loaders matching CDNA5 WMMA VGPR layouts (ISA 7.12.2).
// ---------------------------------------------------------------------------
__device__ __forceinline__ v16h frag_a(const _Float16* base, int lane) {
  const int mrow = lane & 15;
  const int ks   = (lane >> 4) << 3;    // 0 or 8
  const _Float16* p = base + mrow * LDL;
  union { v16h v; v8h h[2]; } u;
  u.h[0] = *(const v8h*)(p + ks);
  u.h[1] = *(const v8h*)(p + 16 + ks);
  return u.v;
}
__device__ __forceinline__ v16h frag_b(const _Float16* base, int lane) {
  const int nrow = lane & 15;
  const int ks   = (lane >> 4) << 3;    // 0 or 8
  const _Float16* p = base + nrow * LDL + ks;
  union { v16h v; v8h h[2]; } u;
  u.h[0] = *(const v8h*)(p);
  u.h[1] = *(const v8h*)(p + 16);
  return u.v;
}

// ---------------------------------------------------------------------------
// Phase 2: out[head] = phiQ(2048x256) * phiK(2048x256)^T, f32 accumulate.
// 128x128 C-tile / block, 8 waves (2x4), 64x32 subtile per wave.
// Double-buffered LDS fed by async global->LDS copies.
// ---------------------------------------------------------------------------
__global__ void __launch_bounds__(256)
rff_gemm_kernel(const _Float16* __restrict__ phiQ,
                const _Float16* __restrict__ phiK,
                float* __restrict__ out)
{
  __shared__ _Float16 As[2][BM * LDL];   // 2 x 10 KB
  __shared__ _Float16 Bs[2][BN * LDL];   // 2 x 10 KB

  const int t = threadIdx.x;
  const int lane = t & 31, wid = t >> 5;
  const int waveM = wid >> 2;
  const int waveN = wid & 3;

  const long head = blockIdx.z;
  const _Float16* Q  = phiQ + head * (long)SEQ * NFEAT;
  const _Float16* Kp = phiK + head * (long)SEQ * NFEAT;
  float* O = out + head * (long)SEQ * SEQ;

  const int mbase = blockIdx.y * BM;
  const int nbase = blockIdx.x * BN;

  // this thread's two 16B chunks per operand tile
  const int r0 = t >> 2,            r1 = (t + 256) >> 2;
  const int c0 = (t & 3) << 3,      c1 = ((t + 256) & 3) << 3;
  const _Float16* gq0 = Q  + (long)(mbase + r0) * NFEAT + c0;
  const _Float16* gq1 = Q  + (long)(mbase + r1) * NFEAT + c1;
  const _Float16* gk0 = Kp + (long)(nbase + r0) * NFEAT + c0;
  const _Float16* gk1 = Kp + (long)(nbase + r1) * NFEAT + c1;
  const unsigned la0 = lds_off(&As[0][r0 * LDL + c0]);
  const unsigned la1 = lds_off(&As[0][r1 * LDL + c1]);
  const unsigned lb0 = lds_off(&Bs[0][r0 * LDL + c0]);
  const unsigned lb1 = lds_off(&Bs[0][r1 * LDL + c1]);

  v8f acc[4][2];
#pragma unroll
  for (int i = 0; i < 4; ++i)
#pragma unroll
    for (int j = 0; j < 2; ++j)
#pragma unroll
      for (int e = 0; e < 8; ++e) acc[i][j][e] = 0.f;

  // prologue: tile 0 -> buffer 0
  async_cp16(la0, gq0);
  async_cp16(la1, gq1);
  async_cp16(lb0, gk0);
  async_cp16(lb1, gk1);

  for (int it = 0; it < NFEAT / BK; ++it) {
    const int cur = it & 1;
    if (it + 1 < NFEAT / BK) {
      const int kb = (it + 1) * BK;               // halves
      const unsigned bo = (unsigned)((cur ^ 1) * BUFBYTES);
      async_cp16(la0 + bo, gq0 + kb);
      async_cp16(la1 + bo, gq1 + kb);
      async_cp16(lb0 + bo, gk0 + kb);
      async_cp16(lb1 + bo, gk1 + kb);
      wait_async_le4();                           // tile `it` landed
    } else {
      wait_async_0();
    }
    __syncthreads();                              // all waves' copies visible

    const _Float16* Abuf = &As[cur][0];
    const _Float16* Bbuf = &Bs[cur][0];
    v16h bf0 = frag_b(Bbuf + (waveN * 32 +  0) * LDL, lane);
    v16h bf1 = frag_b(Bbuf + (waveN * 32 + 16) * LDL, lane);
#pragma unroll
    for (int mi = 0; mi < 4; ++mi) {
      v16h af = frag_a(Abuf + (waveM * 64 + mi * 16) * LDL, lane);
      acc[mi][0] = __builtin_amdgcn_wmma_f32_16x16x32_f16(
          false, af, false, bf0, (short)0, acc[mi][0], false, false);
      acc[mi][1] = __builtin_amdgcn_wmma_f32_16x16x32_f16(
          false, af, false, bf1, (short)0, acc[mi][1], false, false);
    }
    __syncthreads();  // dscnt-drain + barrier: safe to overwrite this buffer
  }

  // C/D layout: VGPR r -> M = r + 8*(lane>=16), N = lane & 15
  const int n0 = lane & 15;
  const int mh = (lane >> 4) << 3;
#pragma unroll
  for (int mi = 0; mi < 4; ++mi) {
    const long gm0 = (long)(mbase + waveM * 64 + mi * 16 + mh);
#pragma unroll
    for (int ni = 0; ni < 2; ++ni) {
      const long gn = nbase + waveN * 32 + ni * 16 + n0;
#pragma unroll
      for (int r = 0; r < 8; ++r)
        __builtin_nontemporal_store(acc[mi][ni][r], &O[(gm0 + r) * SEQ + gn]);
    }
  }
}

// ---------------------------------------------------------------------------
extern "C" void kernel_launch(void* const* d_in, const int* in_sizes, int n_in,
                              void* d_out, int out_size, void* d_ws, size_t ws_size,
                              hipStream_t stream)
{
  (void)in_sizes; (void)n_in; (void)out_size; (void)ws_size;
  const float* q = (const float*)d_in[0];
  const float* k = (const float*)d_in[1];
  const float* W = (const float*)d_in[2];
  const float* b = (const float*)d_in[3];
  float* out = (float*)d_out;

  const long BH    = 4L * 16;
  const long nrows = BH * SEQ;              // 131072 rows per tensor

  _Float16* phiQ = (_Float16*)d_ws;         // 64 MB
  _Float16* phiK = phiQ + nrows * NFEAT;    // 64 MB

  phi_kernel<<<(int)(nrows / PROWS), 256, 0, stream>>>(q, W, b, phiQ);
  phi_kernel<<<(int)(nrows / PROWS), 256, 0, stream>>>(k, W, b, phiK);

  dim3 grid(SEQ / BN, SEQ / BM, (unsigned)BH);
  rff_gemm_kernel<<<grid, 256, 0, stream>>>(phiQ, phiK, out);
}
